// GatedDeltaNetAttention_24507083391404
// MI455X (gfx1250) — matrix operand order repaired and checked by reading, not compiled, not run on hardware
//
#include <hip/hip_runtime.h>
#include <hip/hip_bf16.h>

// ---------------------------------------------------------------------------
// Gated DeltaNet block for MI455X (gfx1250, wave32, WMMA, async global->LDS,
// chunked WY-transform delta rule).
// ---------------------------------------------------------------------------

#define BB 2
#define TT 2048
#define DD 1024
#define HH 16
#define DKK 64
#define MM (BB * TT)          // 4096 rows
#define CONV_K 4
#define CC 32                 // delta-rule chunk length

typedef __bf16 bf16_t;
typedef __attribute__((ext_vector_type(16))) __bf16 v16bf;
typedef __attribute__((ext_vector_type(8)))  __bf16 bf16x8;
typedef __attribute__((ext_vector_type(8)))  float  v8f;
typedef int i32x4 __attribute__((vector_size(16)));

#if __has_builtin(__builtin_amdgcn_global_load_async_to_lds_b128) && \
    __has_builtin(__builtin_amdgcn_s_wait_asynccnt)
#define USE_ASYNC_LDS 1
#else
#define USE_ASYNC_LDS 0
#endif

// ---------------------------------------------------------------------------
// fp32 -> bf16 conversion (grid-stride), row-major preserved
// ---------------------------------------------------------------------------
__global__ void cvt_bf16(const float* __restrict__ in, bf16_t* __restrict__ out, int n)
{
    for (int i = blockIdx.x * blockDim.x + threadIdx.x; i < n; i += gridDim.x * blockDim.x)
        out[i] = (bf16_t)in[i];
}

// ---------------------------------------------------------------------------
// fp32 [K][N] -> bf16 transposed [N][K]
// ---------------------------------------------------------------------------
__global__ void cvt_transpose_bf16(const float* __restrict__ in, bf16_t* __restrict__ out)
{
    const int i = blockIdx.x * blockDim.x + threadIdx.x;
    if (i >= DD * DD) return;
    const int n = i & (DD - 1);
    const int k = i >> 10;
    out[(size_t)n * DD + k] = (bf16_t)in[(size_t)k * DD + n];
}

// ---------------------------------------------------------------------------
// WMMA fragment loaders (wave32; ISA 7.12.2 layouts). Both loaders read
// contiguous bf16 pairs -> compiler merges them into ds_load_b128.
// ---------------------------------------------------------------------------
__device__ __forceinline__ v16bf ldA_rm(const bf16_t* m, int ld, int m0, int k0,
                                        int half, int l16)
{   // A[m][k] from row-major [m][k]
    v16bf a;
    const bf16_t* r = m + (m0 + l16) * ld;
#pragma unroll
    for (int i = 0; i < 8; ++i) {
        const int k2 = (i < 4 ? 2 * i : 16 + 2 * (i - 4)) + half * 8 + k0;
        a[2 * i]     = r[k2];
        a[2 * i + 1] = r[k2 + 1];
    }
    return a;
}
__device__ __forceinline__ v16bf ldB_tr(const bf16_t* m, int ld, int n0, int k0,
                                        int half, int l16)
{   // B[k][n] from Bt-style [n][k] storage
    v16bf b;
    const bf16_t* r = m + (n0 + l16) * ld;
#pragma unroll
    for (int i = 0; i < 8; ++i) {
        const int k2 = (i < 4 ? 2 * i : 16 + 2 * (i - 4)) + half * 8 + k0;
        b[2 * i]     = r[k2];
        b[2 * i + 1] = r[k2 + 1];
    }
    return b;
}
__device__ __forceinline__ v8f vzero8()
{
    v8f z;
#pragma unroll
    for (int e = 0; e < 8; ++e) z[e] = 0.0f;
    return z;
}

// ---------------------------------------------------------------------------
// bf16 GEMM core: C[M,N] = A[M,K] * Bt[N,K]^T, 128x128 tile, async staged.
// ---------------------------------------------------------------------------
#define TILE_M 128
#define TILE_N 128
#define TILE_K 32
#define LDS_PAD 8

typedef bf16_t lds_row_t[TILE_K + LDS_PAD];

__device__ __forceinline__ void async_cp16(const bf16_t* g, bf16_t* l)
{
#if USE_ASYNC_LDS
    typedef __attribute__((address_space(1))) i32x4 gvec_t;
    typedef __attribute__((address_space(3))) i32x4 lvec_t;
    __builtin_amdgcn_global_load_async_to_lds_b128((gvec_t*)g, (lvec_t*)l, 0, 0);
#else
    *(bf16x8*)l = *(const bf16x8*)g;
#endif
}

__device__ __forceinline__ void stage_tiles(const bf16_t* __restrict__ A,
                                            const bf16_t* __restrict__ Bt,
                                            lds_row_t* As, lds_row_t* Bs,
                                            int blockM, int blockN, int kb,
                                            int K, int tid)
{
    const int row    = tid >> 1;
    const int colseg = (tid & 1) * 16;
    const bf16_t* ga = A  + (size_t)(blockM + row) * K + kb + colseg;
    const bf16_t* gb = Bt + (size_t)(blockN + row) * K + kb + colseg;
#if USE_ASYNC_LDS
    async_cp16(ga,     &As[row][colseg]);
    async_cp16(ga + 8, &As[row][colseg + 8]);
    async_cp16(gb,     &Bs[row][colseg]);
    async_cp16(gb + 8, &Bs[row][colseg + 8]);
#else
    bf16x8 a0 = *(const bf16x8*)(ga);
    bf16x8 a1 = *(const bf16x8*)(ga + 8);
    bf16x8 b0 = *(const bf16x8*)(gb);
    bf16x8 b1 = *(const bf16x8*)(gb + 8);
    *(bf16x8*)&As[row][colseg]     = a0;
    *(bf16x8*)&As[row][colseg + 8] = a1;
    *(bf16x8*)&Bs[row][colseg]     = b0;
    *(bf16x8*)&Bs[row][colseg + 8] = b1;
#endif
}

__device__ __forceinline__ void gemm_core(const bf16_t* __restrict__ A,
                                          const bf16_t* __restrict__ Bt,
                                          float* __restrict__ C,
                                          int M, int N, int K)
{
    __shared__ bf16_t As[2][TILE_M][TILE_K + LDS_PAD];
    __shared__ bf16_t Bs[2][TILE_N][TILE_K + LDS_PAD];

    const int tid   = threadIdx.x;
    const int wid   = tid >> 5;
    const int lane  = tid & 31;
    const int half  = lane >> 4;
    const int l16   = lane & 15;
    const int waveM = wid & 3;
    const int waveN = wid >> 2;
    const int blockM = blockIdx.y * TILE_M;
    const int blockN = blockIdx.x * TILE_N;

    v8f acc[2][4];
#pragma unroll
    for (int mi = 0; mi < 2; ++mi)
#pragma unroll
        for (int ni = 0; ni < 4; ++ni) acc[mi][ni] = vzero8();

    stage_tiles(A, Bt, As[0], Bs[0], blockM, blockN, 0, K, tid);

    const int nk = K / TILE_K;
    for (int it = 0; it < nk; ++it) {
        const int cur = it & 1;
        const bool have_next = (it + 1) < nk;
        if (have_next)
            stage_tiles(A, Bt, As[cur ^ 1], Bs[cur ^ 1], blockM, blockN,
                        (it + 1) * TILE_K, K, tid);
#if USE_ASYNC_LDS
        if (have_next) __builtin_amdgcn_s_wait_asynccnt(4);
        else           __builtin_amdgcn_s_wait_asynccnt(0);
#endif
        __syncthreads();

        v16bf afrag[2];
#pragma unroll
        for (int mi = 0; mi < 2; ++mi)
            afrag[mi] = ldA_rm(&As[cur][0][0], TILE_K + LDS_PAD,
                               waveM * 32 + mi * 16, 0, half, l16);
#pragma unroll
        for (int ni = 0; ni < 4; ++ni) {
            v16bf bfrag = ldB_tr(&Bs[cur][0][0], TILE_K + LDS_PAD,
                                 waveN * 64 + ni * 16, 0, half, l16);
#pragma unroll
            for (int mi = 0; mi < 2; ++mi)
                acc[mi][ni] = __builtin_amdgcn_wmma_f32_16x16x32_bf16(
                    false, afrag[mi], false, bfrag, (short)0, acc[mi][ni],
                    false, false);
        }
        __syncthreads();
    }

#pragma unroll
    for (int mi = 0; mi < 2; ++mi)
#pragma unroll
        for (int ni = 0; ni < 4; ++ni) {
            const int gcol = blockN + waveN * 64 + ni * 16 + l16;
#pragma unroll
            for (int r = 0; r < 8; ++r) {
                const int grow = blockM + waveM * 32 + mi * 16 + r + half * 8;
                C[(size_t)grow * N + gcol] = acc[mi][ni][r];
            }
        }
}

__global__ __launch_bounds__(256) void gemm4_bf16_wmma(
    const bf16_t* __restrict__ A,
    const bf16_t* B0, const bf16_t* B1, const bf16_t* B2, const bf16_t* B3,
    float* C0, float* C1, float* C2, float* C3,
    int M, int N, int K)
{
    const int z = blockIdx.z;
    const bf16_t* Bt = (z == 0) ? B0 : (z == 1) ? B1 : (z == 2) ? B2 : B3;
    float*        C  = (z == 0) ? C0 : (z == 1) ? C1 : (z == 2) ? C2 : C3;
    gemm_core(A, Bt, C, M, N, K);
}

__global__ __launch_bounds__(256) void gemm_bf16_wmma(
    const bf16_t* __restrict__ A, const bf16_t* __restrict__ Bt,
    float* __restrict__ C, int M, int N, int K)
{
    gemm_core(A, Bt, C, M, N, K);
}

// ---------------------------------------------------------------------------
// g/beta projections, conv+silu, l2norm
// ---------------------------------------------------------------------------
__global__ void proj_gbeta(const float* __restrict__ x,
                           const float* __restrict__ Wa,
                           const float* __restrict__ Wb,
                           const float* __restrict__ A_log,
                           const float* __restrict__ dt_bias,
                           float* __restrict__ g, float* __restrict__ beta)
{
    const int idx = blockIdx.x * blockDim.x + threadIdx.x;
    if (idx >= MM * HH) return;
    const int h   = idx & (HH - 1);
    const int row = idx >> 4;
    const float* xr = x + (size_t)row * DD;
    float a = 0.0f, b = 0.0f;
    for (int k = 0; k < DD; ++k) {
        const float xv = xr[k];
        a += xv * Wa[k * HH + h];
        b += xv * Wb[k * HH + h];
    }
    const float sp  = a + dt_bias[h];
    const float spv = (sp > 20.0f) ? sp : log1pf(__expf(sp));
    g[idx]    = -__expf(A_log[h]) * spv;
    beta[idx] = 1.0f / (1.0f + __expf(-b));
}

__global__ void conv_silu(const float* __restrict__ in,
                          const float* __restrict__ w,
                          float* __restrict__ out)
{
    const int idx = blockIdx.x * blockDim.x + threadIdx.x;
    if (idx >= MM * DD) return;
    const int c = idx & (DD - 1);
    const int t = (idx >> 10) & (TT - 1);
    float acc = 0.0f;
#pragma unroll
    for (int j = 0; j < CONV_K; ++j) {
        const int dt = j - (CONV_K - 1);
        if (t + dt >= 0) acc += in[idx + dt * DD] * w[c * CONV_K + j];
    }
    out[idx] = acc / (1.0f + __expf(-acc));
}

__global__ void l2norm_scale(float* __restrict__ p, float scale)
{
    const int idx = blockIdx.x * blockDim.x + threadIdx.x;
    if (idx >= MM * HH) return;
    float* v = p + (size_t)idx * DKK;
    float s = 0.0f;
#pragma unroll 8
    for (int d = 0; d < DKK; ++d) s += v[d] * v[d];
    const float r = rsqrtf(s + 1e-6f) * scale;
#pragma unroll 8
    for (int d = 0; d < DKK; ++d) v[d] *= r;
}

// ---------------------------------------------------------------------------
// Chunked gated delta rule (WY/UT transform), chunk C=32, one block per (b,h).
// All LDS operand mirrors are oriented so fragment loads are contiguous:
//   S16t[v][d], U16t[v][t], LK16 = L[t][s] then KdT[d][s].
// ---------------------------------------------------------------------------
#define LDB 72   // bf16 row stride for [.][64] LDS mats
#define LDC 40   // bf16 row stride for [.][32] LDS mats
#define LDM 36   // fp32 row stride for 32x32 M

__global__ __launch_bounds__(256) void chunk_delta(
    const float* __restrict__ q, const float* __restrict__ k,
    const float* __restrict__ v, const float* __restrict__ g,
    const float* __restrict__ beta, float* __restrict__ o)
{
    __shared__ float  Sf[DKK * DKK];        // state S[d][v] (fp32 master)
    __shared__ bf16_t S16t[DKK * LDB];      // bf16 mirror, transposed: [v][d]
    __shared__ bf16_t K16[CC * LDB];        // k chunk [t][d]
    __shared__ bf16_t sc16[CC * LDB];       // scaled k: beta*k -> beta*e^G*k
    __shared__ bf16_t Q16[CC * LDB];        // q chunk [t][d]
    __shared__ bf16_t U16t[DKK * LDC];      // U transposed: [v][t]
    __shared__ bf16_t LK16[DKK * LDC];      // L[t][s] (rows 0..31) then KdT[d][s]
    __shared__ float  Vf[CC * DKK];         // V -> R -> QS0
    __shared__ float  Mf[CC * LDM];         // decay-masked beta KK^T (strict lower)
    __shared__ float  Gc[CC], eG[CC], bv[CC];

    const int bh  = blockIdx.x;
    const int b   = bh / HH;
    const int h   = bh % HH;
    const int tid = threadIdx.x;
    const int wid  = tid >> 5;
    const int lane = tid & 31;
    const int half = lane >> 4;
    const int l16  = lane & 15;
    const size_t strideT = (size_t)HH * DKK;

    for (int i = tid; i < DKK * DKK; i += 256) {
        Sf[i] = 0.0f;
        S16t[(i >> 6) * LDB + (i & 63)] = (bf16_t)0.0f;
    }
    __syncthreads();

    for (int c = 0; c < TT / CC; ++c) {
        const int t0 = c * CC;
        // ---- p0: per-step scalars ------------------------------------------
        if (tid < CC) {
            const size_t gi = ((size_t)b * TT + t0 + tid) * HH + h;
            Gc[tid] = g[gi];
            bv[tid] = beta[gi];
        }
        __syncthreads();
        // ---- p1: load chunk; cumsum ----------------------------------------
        for (int i = tid; i < CC * DKK; i += 256) {
            const int t = i >> 6, d = i & 63;
            const size_t gb = ((size_t)b * TT + t0 + t) * strideT + (size_t)h * DKK + d;
            const float kv = k[gb];
            K16[t * LDB + d]  = (bf16_t)kv;
            sc16[t * LDB + d] = (bf16_t)(kv * bv[t]);     // beta*k
            Q16[t * LDB + d]  = (bf16_t)q[gb];
            Vf[t * DKK + d]   = v[gb];
        }
        if (tid == 0) {
            float a = 0.0f;
#pragma unroll
            for (int t = 0; t < CC; ++t) { a += Gc[t]; Gc[t] = a; }
        }
        __syncthreads();
        // ---- p2: eG; G1: Mf = mask(beta KK^T) ------------------------------
        if (tid < CC) eG[tid] = __expf(Gc[tid]);
        if (wid < 4) {
            const int m0 = (wid >> 1) * 16, n0 = (wid & 1) * 16;
            v8f acc = vzero8();
#pragma unroll
            for (int ks = 0; ks < 2; ++ks)
                acc = __builtin_amdgcn_wmma_f32_16x16x32_bf16(
                    false, ldA_rm(sc16, LDB, m0, ks * 32, half, l16),
                    false, ldB_tr(K16, LDB, n0, ks * 32, half, l16),
                    (short)0, acc, false, false);
#pragma unroll
            for (int r = 0; r < 8; ++r) {
                const int tr = m0 + r + half * 8, s = n0 + l16;
                Mf[tr * LDM + s] = (s < tr) ? acc[r] * __expf(Gc[tr] - Gc[s]) : 0.0f;
            }
        }
        __syncthreads();
        // ---- p3: sc16 = beta*e^G*k -----------------------------------------
        for (int i = tid; i < CC * DKK; i += 256) {
            const int t = i >> 6, d = i & 63;
            const size_t gb = ((size_t)b * TT + t0 + t) * strideT + (size_t)h * DKK + d;
            sc16[t * LDB + d] = (bf16_t)(k[gb] * bv[t] * eG[t]);
        }
        __syncthreads();
        // ---- p4: G2: Vf = beta*V - (beta e^G K) S0  (= R) ------------------
        {
            const int m0 = (wid >> 2) * 16, n0 = (wid & 3) * 16;
            v8f acc = vzero8();
#pragma unroll
            for (int ks = 0; ks < 2; ++ks)
                acc = __builtin_amdgcn_wmma_f32_16x16x32_bf16(
                    false, ldA_rm(sc16, LDB, m0, ks * 32, half, l16),
                    false, ldB_tr(S16t, LDB, n0, ks * 32, half, l16),
                    (short)0, acc, false, false);
#pragma unroll
            for (int r = 0; r < 8; ++r) {
                const int tr = m0 + r + half * 8, vv = n0 + l16;
                Vf[tr * DKK + vv] = bv[tr] * Vf[tr * DKK + vv] - acc[r];
            }
        }
        __syncthreads();
        // ---- p5: forward substitution (lane-private column) ----------------
        if (tid < DKK) {
            float Ucol[CC];
#pragma unroll
            for (int t = 0; t < CC; ++t) {
                float u = Vf[t * DKK + tid];
#pragma unroll
                for (int s = 0; s < t; ++s) u -= Mf[t * LDM + s] * Ucol[s];
                Ucol[t] = u;
                U16t[tid * LDC + t] = (bf16_t)u;
            }
        }
        __syncthreads();
        // ---- p6: G4: Vf = Q S0 ; G3: LK16 = mask(Q K^T) --------------------
        {
            const int m0 = (wid >> 2) * 16, n0 = (wid & 3) * 16;
            v8f acc = vzero8();
#pragma unroll
            for (int ks = 0; ks < 2; ++ks)
                acc = __builtin_amdgcn_wmma_f32_16x16x32_bf16(
                    false, ldA_rm(Q16, LDB, m0, ks * 32, half, l16),
                    false, ldB_tr(S16t, LDB, n0, ks * 32, half, l16),
                    (short)0, acc, false, false);
#pragma unroll
            for (int r = 0; r < 8; ++r) {
                const int tr = m0 + r + half * 8, vv = n0 + l16;
                Vf[tr * DKK + vv] = acc[r];
            }
        }
        if (wid < 4) {
            const int m0 = (wid >> 1) * 16, n0 = (wid & 1) * 16;
            v8f acc = vzero8();
#pragma unroll
            for (int ks = 0; ks < 2; ++ks)
                acc = __builtin_amdgcn_wmma_f32_16x16x32_bf16(
                    false, ldA_rm(Q16, LDB, m0, ks * 32, half, l16),
                    false, ldB_tr(K16, LDB, n0, ks * 32, half, l16),
                    (short)0, acc, false, false);
#pragma unroll
            for (int r = 0; r < 8; ++r) {
                const int tr = m0 + r + half * 8, s = n0 + l16;
                LK16[tr * LDC + s] = (s <= tr)
                    ? (bf16_t)(acc[r] * __expf(Gc[tr] - Gc[s])) : (bf16_t)0.0f;
            }
        }
        __syncthreads();
        // ---- p7: G5: O = e^G * (Q S0) + L U -> global ----------------------
        {
            const int m0 = (wid >> 2) * 16, n0 = (wid & 3) * 16;
            v8f acc = __builtin_amdgcn_wmma_f32_16x16x32_bf16(
                false, ldA_rm(LK16, LDC, m0, 0, half, l16),
                false, ldB_tr(U16t, LDC, n0, 0, half, l16),
                (short)0, vzero8(), false, false);
#pragma unroll
            for (int r = 0; r < 8; ++r) {
                const int tr = m0 + r + half * 8, vv = n0 + l16;
                const size_t go = ((size_t)b * TT + t0 + tr) * strideT
                                + (size_t)h * DKK + vv;
                o[go] = eG[tr] * Vf[tr * DKK + vv] + acc[r];
            }
        }
        __syncthreads();   // L reads done before LK16 is rewritten as KdT
        // ---- p7b: LK16 = KdT[d][s] = e^{GC-Gs} k[s][d] ---------------------
        for (int i = tid; i < CC * DKK; i += 256) {
            const int t = i >> 6, d = i & 63;
            const size_t gb = ((size_t)b * TT + t0 + t) * strideT + (size_t)h * DKK + d;
            LK16[d * LDC + t] = (bf16_t)(k[gb] * __expf(Gc[CC - 1] - Gc[t]));
        }
        __syncthreads();
        // ---- p8: G6: S = e^{GC} S + Kd^T U ---------------------------------
        {
            const float eGC = eG[CC - 1];
            const int m0 = (wid >> 1) * 16;
#pragma unroll
            for (int j = 0; j < 2; ++j) {
                const int n0 = ((wid & 1) * 2 + j) * 16;
                v8f acc = __builtin_amdgcn_wmma_f32_16x16x32_bf16(
                    false, ldA_rm(LK16, LDC, m0, 0, half, l16),
                    false, ldB_tr(U16t, LDC, n0, 0, half, l16),
                    (short)0, vzero8(), false, false);
#pragma unroll
                for (int r = 0; r < 8; ++r) {
                    const int dr = m0 + r + half * 8, vv = n0 + l16;
                    const float ns = eGC * Sf[dr * DKK + vv] + acc[r];
                    Sf[dr * DKK + vv]   = ns;
                    S16t[vv * LDB + dr] = (bf16_t)ns;
                }
            }
        }
        __syncthreads();
    }
}

// ---------------------------------------------------------------------------
// Gated RMSNorm -> bf16
// ---------------------------------------------------------------------------
__global__ void rmsnorm_gate(const float* __restrict__ o,
                             const float* __restrict__ gatepre,
                             const float* __restrict__ rms_w,
                             bf16_t* __restrict__ ob)
{
    const int idx = blockIdx.x * blockDim.x + threadIdx.x;
    if (idx >= MM * HH) return;
    const float* op = o + (size_t)idx * DKK;
    const float* gp = gatepre + (size_t)idx * DKK;
    float s = 0.0f;
#pragma unroll 8
    for (int d = 0; d < DKK; ++d) s += op[d] * op[d];
    const float r = rsqrtf(s * (1.0f / DKK) + 1e-5f);
#pragma unroll 8
    for (int d = 0; d < DKK; ++d) {
        float val = op[d] * r * rms_w[d];
        const float gt = gp[d];
        val *= gt / (1.0f + __expf(-gt));
        ob[(size_t)idx * DKK + d] = (bf16_t)val;
    }
}

// ---------------------------------------------------------------------------
// Host-side orchestration
// ---------------------------------------------------------------------------
extern "C" void kernel_launch(void* const* d_in, const int* in_sizes, int n_in,
                              void* d_out, int out_size, void* d_ws, size_t ws_size,
                              hipStream_t stream)
{
    (void)in_sizes; (void)n_in; (void)out_size; (void)ws_size;

    const float* x       = (const float*)d_in[0];
    const float* Wq      = (const float*)d_in[1];
    const float* Wk      = (const float*)d_in[2];
    const float* Wv      = (const float*)d_in[3];
    const float* Wa      = (const float*)d_in[4];
    const float* Wb      = (const float*)d_in[5];
    const float* Wg      = (const float*)d_in[6];
    const float* Wo      = (const float*)d_in[7];
    const float* conv_q  = (const float*)d_in[8];
    const float* conv_k  = (const float*)d_in[9];
    const float* conv_v  = (const float*)d_in[10];
    const float* A_log   = (const float*)d_in[11];
    const float* dt_bias = (const float*)d_in[12];
    const float* rms_w   = (const float*)d_in[13];
    float* out = (float*)d_out;

    char* ws = (char*)d_ws;
    size_t off = 0;
    auto bump = [&](size_t bytes) {
        char* p = ws + off;
        off = (off + bytes + 255) & ~(size_t)255;
        return p;
    };
    const size_t NE = (size_t)MM * DD;
    bf16_t* xb   = (bf16_t*)bump(NE * sizeof(bf16_t));
    bf16_t* wqT  = (bf16_t*)bump((size_t)DD * DD * sizeof(bf16_t));
    bf16_t* wkT  = (bf16_t*)bump((size_t)DD * DD * sizeof(bf16_t));
    bf16_t* wvT  = (bf16_t*)bump((size_t)DD * DD * sizeof(bf16_t));
    bf16_t* wgT  = (bf16_t*)bump((size_t)DD * DD * sizeof(bf16_t));
    bf16_t* woT  = (bf16_t*)bump((size_t)DD * DD * sizeof(bf16_t));
    float*  qpre = (float*)bump(NE * sizeof(float));
    float*  kpre = (float*)bump(NE * sizeof(float));
    float*  vpre = (float*)bump(NE * sizeof(float));
    float*  gpre = (float*)bump(NE * sizeof(float));
    float*  qbuf = (float*)bump(NE * sizeof(float));
    float*  kbuf = (float*)bump(NE * sizeof(float));
    float*  vbuf = (float*)bump(NE * sizeof(float));
    float*  gdec = (float*)bump((size_t)MM * HH * sizeof(float));
    float*  bdec = (float*)bump((size_t)MM * HH * sizeof(float));
    float*  osc  = (float*)bump(NE * sizeof(float));
    bf16_t* ob   = (bf16_t*)bump(NE * sizeof(bf16_t));

    const int tpGrid = (DD * DD + 255) / 256;

    cvt_bf16<<<1024, 256, 0, stream>>>(x, xb, (int)NE);
    cvt_transpose_bf16<<<tpGrid, 256, 0, stream>>>(Wq, wqT);
    cvt_transpose_bf16<<<tpGrid, 256, 0, stream>>>(Wk, wkT);
    cvt_transpose_bf16<<<tpGrid, 256, 0, stream>>>(Wv, wvT);
    cvt_transpose_bf16<<<tpGrid, 256, 0, stream>>>(Wg, wgT);
    cvt_transpose_bf16<<<tpGrid, 256, 0, stream>>>(Wo, woT);

    const dim3 gemmGrid4(DD / TILE_N, MM / TILE_M, 4);
    gemm4_bf16_wmma<<<gemmGrid4, 256, 0, stream>>>(
        xb, wqT, wkT, wvT, wgT, qpre, kpre, vpre, gpre, MM, DD, DD);

    proj_gbeta<<<(MM * HH + 255) / 256, 256, 0, stream>>>(x, Wa, Wb, A_log, dt_bias,
                                                          gdec, bdec);

    const int convBlocks = (int)((NE + 255) / 256);
    conv_silu<<<convBlocks, 256, 0, stream>>>(qpre, conv_q, qbuf);
    conv_silu<<<convBlocks, 256, 0, stream>>>(kpre, conv_k, kbuf);
    conv_silu<<<convBlocks, 256, 0, stream>>>(vpre, conv_v, vbuf);

    const int nHeads = MM * HH;
    l2norm_scale<<<(nHeads + 255) / 256, 256, 0, stream>>>(qbuf, 0.125f);
    l2norm_scale<<<(nHeads + 255) / 256, 256, 0, stream>>>(kbuf, 1.0f);

    chunk_delta<<<BB * HH, 256, 0, stream>>>(qbuf, kbuf, vbuf, gdec, bdec, osc);

    rmsnorm_gate<<<(nHeads + 255) / 256, 256, 0, stream>>>(osc, gpre, rms_w, ob);

    const dim3 gemmGrid(DD / TILE_N, MM / TILE_M, 1);
    gemm_bf16_wmma<<<gemmGrid, 256, 0, stream>>>(ob, woT, out, MM, DD, DD);
}
